// spk_vq_vae_densenet_50294067036783
// MI455X (gfx1250) — matrix operand order repaired and checked by reading, not compiled, not run on hardware
//
#include <hip/hip_runtime.h>

// ---------------------------------------------------------------------------
// VQ-VAE (DenseNet encoder + VQ + deconv decoder) for gfx1250 (MI455X).
// All convs are GEMMs on v_wmma_f32_16x16x32_bf16 (bf16 A/B, f32 accum).
// Layouts:
//   activations: row-major [N, Cpad], N = B*L with n = b*L + l
//   A_pre (bf16): [N, Kpad] (Kpad mult of 32, zero padded), built by fused
//                 BN-ReLU (+upsample) (+im2col k=3) kernels
//   weights (bf16): [Npad, Kpad] row-major = B^T, so B-fragments are
//                   two 16-byte row reads per lane (matches ISA B VGPR layout)
// GEMM: each wave owns one 16-row M tile and an NT*16-wide N strip, loading
// the A fragment once per K-step and reusing it across NT WMMAs.
// Input index map (jax tree_leaves order = sorted dict keys):
//   blocks 1..3: layer i -> base = blk*60 + i*6 : [b1,b2,g1,g2,w1,w2]
//   180: conv1_w ; 181..194: decoder [b1,b2,b3,dc0_b,dc0_w,dc1_b,dc1_w,
//        dc2_b,dc2_w,dc3_b,dc3_w,g1,g2,g3] ; 195: embed ;
//   196..198 trans1 [b,g,w]; 199..201 trans2; 202..204 trans3; 205: x
// ---------------------------------------------------------------------------

typedef __attribute__((ext_vector_type(16))) __bf16 v16bf;
typedef __attribute__((ext_vector_type(8)))  float  v8f;
typedef __attribute__((ext_vector_type(4)))  unsigned int v4u;

union BFrag { v16bf v; v4u q[2]; };

// ------ WMMA GEMM: Out[M,Nvalid] = A[M,K] * W[N,K]^T (+bias), NT N-tiles ----
template <int NT>
__global__ __launch_bounds__(256)
void k_gemm_nt(const __bf16* __restrict__ A, int lda,
               const __bf16* __restrict__ W, int ldw,
               float* __restrict__ Out, int ldo,
               const float* __restrict__ bias,
               int M, int K, int Nvalid)
{
  const int lane  = threadIdx.x & 31;
  const int wave  = threadIdx.x >> 5;
  const int mTile = blockIdx.x * 8 + wave;    // 8 waves per block along M
  const int mBase = mTile << 4;
  if (mBase >= M) return;                     // wave-uniform exit (EXEC stays full)
  const int nBase0 = blockIdx.y * (16 * NT);

  const int hl  = lane >> 4;                  // half-wave select
  const int aK0 = hl << 3;                    // 0 or 8  (A K-offset)
  const int bK0 = hl << 3;                    // 0 or 8  (B K-offset)
  const __bf16* aPtr = A + (size_t)(mBase  + (lane & 15)) * lda + aK0;
  const __bf16* bPtr = W + (size_t)(nBase0 + (lane & 15)) * ldw + bK0;

  v8f acc[NT];
  {
    v8f z = {};
#pragma unroll
    for (int t = 0; t < NT; ++t) acc[t] = z;
  }

  for (int k = 0; k < K; k += 32) {
    BFrag a;
    a.q[0] = *(const v4u*)(aPtr + k);         // K = k+aK0 .. +7
    a.q[1] = *(const v4u*)(aPtr + k + 16);    // K = k+aK0+16 .. +23
#pragma unroll
    for (int t = 0; t < NT; ++t) {
      BFrag b;
      const __bf16* bp = bPtr + (size_t)(t * 16) * ldw + k;
      b.q[0] = *(const v4u*)(bp);             // K = k+bK0 .. +7
      b.q[1] = *(const v4u*)(bp + 16);        // K = k+bK0+16 .. +23
      acc[t] = __builtin_amdgcn_wmma_f32_16x16x32_bf16(
          false, a.v, false, b.v, (short)0, acc[t], false, false);
    }
  }

  const int mOff = mBase + (hl << 3);
#pragma unroll
  for (int t = 0; t < NT; ++t) {
    const int col = nBase0 + t * 16 + (lane & 15);
    if (col < Nvalid) {
      const float bv = bias ? bias[col] : 0.0f;
#pragma unroll
      for (int v = 0; v < 8; ++v)
        Out[(size_t)(mOff + v) * ldo + col] = acc[t][v] + bv;
    }
  }
}

// ---------------- BN stats, deterministic two-stage -------------------------
__global__ void k_bnstats1(const float* __restrict__ src, int lds, int C,
                           int rows, int rpb, float* __restrict__ partial)
{
  int c = threadIdx.x;
  if (c >= C) return;
  int r0 = blockIdx.x * rpb;
  int r1 = r0 + rpb; if (r1 > rows) r1 = rows;
  float s = 0.f, s2 = 0.f;
  for (int r = r0; r < r1; ++r) {
    float v = src[(size_t)r * lds + c];
    s += v; s2 += v * v;
  }
  partial[blockIdx.x * 512 + c]       = s;
  partial[blockIdx.x * 512 + 256 + c] = s2;
}

__global__ void k_bnstats2(const float* __restrict__ partial, int NB, int C,
                           float invN, const float* __restrict__ g,
                           const float* __restrict__ b,
                           float* __restrict__ scale, float* __restrict__ shift)
{
  int c = threadIdx.x;
  if (c >= C) return;
  float s = 0.f, s2 = 0.f;
  for (int i = 0; i < NB; ++i) { s += partial[i*512 + c]; s2 += partial[i*512 + 256 + c]; }
  float mean = s * invN;
  float var  = s2 * invN - mean * mean;
  float rstd = rsqrtf(var + 1e-5f);
  float sc   = g[c] * rstd;
  scale[c] = sc;
  shift[c] = b[c] - mean * sc;
}

// ---------------- fused BN-ReLU -> bf16 (1x1 conv A-matrix) -----------------
__global__ void k_bnrelu_bf16(const float* __restrict__ src, int lds, int C,
                              int Kpad, int rows, const float* __restrict__ scale,
                              const float* __restrict__ shift, __bf16* __restrict__ dst)
{
  size_t idx = (size_t)blockIdx.x * blockDim.x + threadIdx.x;
  if (idx >= (size_t)rows * Kpad) return;
  int c = (int)(idx % Kpad);
  size_t n = idx / Kpad;
  float v = 0.f;
  if (c < C) {
    v = src[n * lds + c] * scale[c] + shift[c];
    v = v > 0.f ? v : 0.f;
  }
  dst[idx] = (__bf16)v;
}

// -------- fused (BN-ReLU)(upsample x2) im2col k=3 -> bf16 A-matrix ----------
// A[n, dl*Cin+ci] = f(src[b*Lsrc + map(l+dl-1), ci]); map = /2 if Lsrc<Lout
__global__ void k_im2col3(const float* __restrict__ src, int lds, int Cin,
                          int Lout, int Lsrc, int Kpad, int rows,
                          const float* __restrict__ scale,
                          const float* __restrict__ shift, int bnrelu,
                          __bf16* __restrict__ dst)
{
  size_t idx = (size_t)blockIdx.x * blockDim.x + threadIdx.x;
  if (idx >= (size_t)rows * Kpad) return;
  int j = (int)(idx % Kpad);
  size_t n = idx / Kpad;
  float v = 0.f;
  if (j < 3 * Cin) {
    int dl = j / Cin, ci = j - dl * Cin;
    int l = (int)(n % Lout), b = (int)(n / Lout);
    int p = l + dl - 1;
    if (p >= 0 && p < Lout) {
      int sp = (Lsrc == Lout) ? p : (p >> 1);
      v = src[((size_t)b * Lsrc + sp) * lds + ci];
      if (bnrelu) { v = v * scale[ci] + shift[ci]; v = v > 0.f ? v : 0.f; }
    }
  }
  dst[idx] = (__bf16)v;
}

// ---------------- avg-pool k=2 s=2 over length ------------------------------
__global__ void k_avgpool2(const float* __restrict__ src, int lds, int C,
                           int Lin, int rowsOut, float* __restrict__ dst, int ldd)
{
  size_t idx = (size_t)blockIdx.x * blockDim.x + threadIdx.x;
  if (idx >= (size_t)rowsOut * C) return;
  int c = (int)(idx % C);
  int n = (int)(idx / C);
  int Lh = Lin >> 1;
  int l = n % Lh, b = n / Lh;
  size_t s = ((size_t)b * Lin + 2 * l) * lds + c;
  dst[(size_t)n * ldd + c] = 0.5f * (src[s] + src[s + lds]);
}

// ---------------- conv1: 1 -> 24 channels, k=3, pad 1 (tiny, VALU) ----------
__global__ void k_conv1(const float* __restrict__ x, const float* __restrict__ w,
                        float* __restrict__ act, int ldd)
{
  size_t idx = (size_t)blockIdx.x * blockDim.x + threadIdx.x;
  if (idx >= (size_t)131072 * 24) return;
  int co = (int)(idx % 24);
  int n  = (int)(idx / 24);
  int l = n & 63;
  float s = 0.f;
#pragma unroll
  for (int dl = 0; dl < 3; ++dl) {
    int p = l + dl - 1;
    if (p >= 0 && p < 64) s += x[n + dl - 1] * w[co * 3 + dl];
  }
  act[(size_t)n * ldd + co] = s;
}

// ---------------- weight conversion to padded bf16 [Npad,Kpad] --------------
// conv weight (Co, Ci, Kt): dst[co*Kpad + kt*Ci + ci] = src[(co*Ci+ci)*Kt + kt]
__global__ void k_wconv(const float* __restrict__ src, int Co, int Ci, int Kt,
                        int Npad, int Kpad, __bf16* __restrict__ dst)
{
  size_t idx = (size_t)blockIdx.x * blockDim.x + threadIdx.x;
  if (idx >= (size_t)Npad * Kpad) return;
  int kk = (int)(idx % Kpad);
  int co = (int)(idx / Kpad);
  float v = 0.f;
  if (co < Co && kk < Kt * Ci) {
    int kt = kk / Ci, ci = kk - kt * Ci;
    v = src[((size_t)co * Ci + ci) * Kt + kt];
  }
  dst[idx] = (__bf16)v;
}

// deconv weight (Ci, Co, 3): dst[co*Kpad + dl*Ci + ci] = src[(ci*Co+co)*3 + (2-dl)]
__global__ void k_wdeconv(const float* __restrict__ src, int Ci, int Co,
                          int Npad, int Kpad, __bf16* __restrict__ dst)
{
  size_t idx = (size_t)blockIdx.x * blockDim.x + threadIdx.x;
  if (idx >= (size_t)Npad * Kpad) return;
  int kk = (int)(idx % Kpad);
  int co = (int)(idx / Kpad);
  float v = 0.f;
  if (co < Co && kk < 3 * Ci) {
    int dl = kk / Ci, ci = kk - dl * Ci;
    v = src[((size_t)ci * Co + co) * 3 + (2 - dl)];
  }
  dst[idx] = (__bf16)v;
}

// ---------------- VQ: argmin over 128 codes (dim=8 along length) ------------
__global__ void k_vq(const float* __restrict__ z, int ldz,
                     const float* __restrict__ embed,
                     int* __restrict__ jout, float* __restrict__ lossPartial)
{
  __shared__ float cb[128 * 8];
  __shared__ float ww[128];
  __shared__ float red[256];
  int tid = threadIdx.x;
  for (int i = tid; i < 1024; i += 256) cb[i] = embed[i];
  __syncthreads();
  if (tid < 128) {
    float s = 0.f;
    for (int d = 0; d < 8; ++d) { float w = cb[tid * 8 + d]; s += w * w; }
    ww[tid] = s;
  }
  __syncthreads();
  int idx = blockIdx.x * 256 + tid;          // (b, ch) pairs, total 2048*216
  float loss = 0.f;
  if (idx < 2048 * 216) {
    int b = idx / 216, ch = idx - b * 216;
    float zl[8]; float zz = 0.f;
#pragma unroll
    for (int d = 0; d < 8; ++d) {
      float v = z[((size_t)b * 8 + d) * ldz + ch];
      zl[d] = v; zz += v * v;
    }
    float best = 3.4e38f; int bj = 0;
    for (int q = 0; q < 128; ++q) {
      float dot = 0.f;
#pragma unroll
      for (int d = 0; d < 8; ++d) dot += zl[d] * cb[q * 8 + d];
      float dist = zz - 2.f * dot + ww[q];   // expanded cdist, first-min wins
      if (dist < best) { best = dist; bj = q; }
    }
    jout[idx] = bj;
#pragma unroll
    for (int d = 0; d < 8; ++d) { float df = zl[d] - cb[bj * 8 + d]; loss += df * df; }
  }
  red[tid] = loss; __syncthreads();
  for (int s = 128; s > 0; s >>= 1) { if (tid < s) red[tid] += red[tid + s]; __syncthreads(); }
  if (tid == 0) lossPartial[blockIdx.x] = red[0];
}

__global__ void k_lossfin(const float* __restrict__ partial, int nb, float invN,
                          float* __restrict__ out)
{
  __shared__ float red[256];
  float s = 0.f;
  for (int i = threadIdx.x; i < nb; i += 256) s += partial[i];
  red[threadIdx.x] = s; __syncthreads();
  for (int st = 128; st > 0; st >>= 1) {
    if (threadIdx.x < st) red[threadIdx.x] += red[threadIdx.x + st];
    __syncthreads();
  }
  if (threadIdx.x == 0) { out[0] = red[0] * invN; out[1] = red[0] * invN; }
}

// ------------- decoder input: gathered codes, nearest-upsampled to L=16 -----
__global__ void k_decin(const float* __restrict__ embed, const int* __restrict__ j,
                        float* __restrict__ dst, int ldd)
{
  size_t idx = (size_t)blockIdx.x * blockDim.x + threadIdx.x;
  if (idx >= (size_t)32768 * 216) return;
  int ch = (int)(idx % 216);
  int n  = (int)(idx / 216);
  int l = n & 15, b = n >> 4;
  int code = j[b * 216 + ch];
  dst[(size_t)n * ldd + ch] = embed[code * 8 + (l >> 1)];
}

// ===========================================================================
extern "C" void kernel_launch(void* const* d_in, const int* in_sizes, int n_in,
                              void* d_out, int out_size, void* d_ws, size_t ws_size,
                              hipStream_t stream)
{
  (void)in_sizes; (void)n_in; (void)out_size; (void)ws_size;
  auto F = [&](int i) { return (const float*)d_in[i]; };
  auto ceilDiv = [](size_t a, size_t b) { return (int)((a + b - 1) / b); };
  auto a32 = [](int c) { return (c + 31) & ~31; };
  auto a16 = [](int c) { return (c + 15) & ~15; };

  char* ws = (char*)d_ws;
  float*  ACT   = (float*) (ws);                          // 80 MB acts
  float*  TMP   = (float*) (ws + ((size_t)80  << 20));    // 40 MB gemm out
  __bf16* APRE  = (__bf16*)(ws + ((size_t)120 << 20));    // 64 MB bf16 A
  __bf16* WB    = (__bf16*)(ws + ((size_t)184 << 20));    // 2  MB bf16 weights
  float*  PART  = (float*) (ws + ((size_t)186 << 20));    // BN partials
  float*  SCALE = (float*) (ws + ((size_t)187 << 20));
  float*  SHIFT = SCALE + 256;
  int*    VQJ   = (int*)   (ws + ((size_t)188 << 20));
  float*  LP    = (float*) (ws + ((size_t)190 << 20));
  float*  outF  = (float*)d_out;

  const int B = 2048, NB = 256;

  // ---- convert all weights to padded bf16 ----
  size_t wOff = 0;
  auto wslot = [&](size_t elems) { __bf16* p = WB + wOff; wOff += (elems + 127) & ~(size_t)127; return p; };

  __bf16 *W1b[30], *W2b[30], *Wtb[3], *Wd[4];
  const int cIn0[3] = {24, 72, 96};
  for (int blk = 0; blk < 3; ++blk)
    for (int i = 0; i < 10; ++i) {
      int base = blk * 60 + i * 6;
      int c = cIn0[blk] + 12 * i;
      int Kp = a32(c), li = blk * 10 + i;
      W1b[li] = wslot((size_t)48 * Kp);
      k_wconv<<<ceilDiv((size_t)48 * Kp, 256), 256, 0, stream>>>(F(base + 4), 48, c, 1, 48, Kp, W1b[li]);
      W2b[li] = wslot((size_t)16 * 160);
      k_wconv<<<ceilDiv((size_t)16 * 160, 256), 256, 0, stream>>>(F(base + 5), 12, 48, 3, 16, 160, W2b[li]);
    }
  const int tCo[3] = {72, 96, 216}, tCi[3] = {144, 192, 216}, tW[3] = {198, 201, 204};
  for (int t = 0; t < 3; ++t) {
    int Np = a16(tCo[t]), Kp = a32(tCi[t]);
    Wtb[t] = wslot((size_t)Np * Kp);
    k_wconv<<<ceilDiv((size_t)Np * Kp, 256), 256, 0, stream>>>(F(tW[t]), tCo[t], tCi[t], 1, Np, Kp, Wtb[t]);
  }
  Wd[0] = wslot((size_t)96 * 672);
  k_wdeconv<<<ceilDiv((size_t)96 * 672, 256), 256, 0, stream>>>(F(191), 216, 96, 96, 672, Wd[0]);
  Wd[1] = wslot((size_t)80 * 288);
  k_wdeconv<<<ceilDiv((size_t)80 * 288, 256), 256, 0, stream>>>(F(189), 96, 72, 80, 288, Wd[1]);
  Wd[2] = wslot((size_t)32 * 224);
  k_wdeconv<<<ceilDiv((size_t)32 * 224, 256), 256, 0, stream>>>(F(187), 72, 24, 32, 224, Wd[2]);
  Wd[3] = wslot((size_t)16 * 96);
  k_wdeconv<<<ceilDiv((size_t)16 * 96, 256), 256, 0, stream>>>(F(185), 24, 1, 16, 96, Wd[3]);

  // ---- helpers ----
  auto bnstats = [&](const float* src, int lds, int C, int rows,
                     const float* g, const float* b) {
    int rpb = ceilDiv((size_t)rows, NB);
    k_bnstats1<<<NB, 256, 0, stream>>>(src, lds, C, rows, rpb, PART);
    k_bnstats2<<<1, 256, 0, stream>>>(PART, NB, C, 1.0f / (float)rows, g, b, SCALE, SHIFT);
  };
  auto gemm = [&](const __bf16* A, int Kp, const __bf16* W, float* Out, int ldo,
                  const float* bias, int M, int Np, int Nv) {
    int ntiles = Np / 16;
    int nt = 1;
    if      (ntiles % 7 == 0) nt = 7;
    else if (ntiles % 6 == 0) nt = 6;
    else if (ntiles % 5 == 0) nt = 5;
    else if (ntiles % 3 == 0) nt = 3;
    else if (ntiles % 2 == 0) nt = 2;
    dim3 grid(ceilDiv((size_t)(M / 16), 8), ntiles / nt);
    switch (nt) {
      case 7: k_gemm_nt<7><<<grid, 256, 0, stream>>>(A, Kp, W, Kp, Out, ldo, bias, M, Kp, Nv); break;
      case 6: k_gemm_nt<6><<<grid, 256, 0, stream>>>(A, Kp, W, Kp, Out, ldo, bias, M, Kp, Nv); break;
      case 5: k_gemm_nt<5><<<grid, 256, 0, stream>>>(A, Kp, W, Kp, Out, ldo, bias, M, Kp, Nv); break;
      case 3: k_gemm_nt<3><<<grid, 256, 0, stream>>>(A, Kp, W, Kp, Out, ldo, bias, M, Kp, Nv); break;
      case 2: k_gemm_nt<2><<<grid, 256, 0, stream>>>(A, Kp, W, Kp, Out, ldo, bias, M, Kp, Nv); break;
      default: k_gemm_nt<1><<<grid, 256, 0, stream>>>(A, Kp, W, Kp, Out, ldo, bias, M, Kp, Nv); break;
    }
  };

  // ---- encoder ----
  const float* x = F(205);
  k_conv1<<<ceilDiv((size_t)131072 * 24, 256), 256, 0, stream>>>(x, F(180), ACT, 144);

  const int CPs[3] = {144, 192, 216};
  const int tb[3] = {196, 199, 202};
  int L = 64, rows = B * 64;
  for (int blk = 0; blk < 3; ++blk) {
    int CP = CPs[blk];
    int c = cIn0[blk];
    for (int i = 0; i < 10; ++i) {
      int base = blk * 60 + i * 6;
      int li = blk * 10 + i;
      int Kp = a32(c);
      bnstats(ACT, CP, c, rows, F(base + 2), F(base + 0));                  // BN1
      k_bnrelu_bf16<<<ceilDiv((size_t)rows * Kp, 256), 256, 0, stream>>>(
          ACT, CP, c, Kp, rows, SCALE, SHIFT, APRE);
      gemm(APRE, Kp, W1b[li], TMP, 48, nullptr, rows, 48, 48);              // 1x1 -> 48
      bnstats(TMP, 48, 48, rows, F(base + 3), F(base + 1));                 // BN2
      k_im2col3<<<ceilDiv((size_t)rows * 160, 256), 256, 0, stream>>>(
          TMP, 48, 48, L, L, 160, rows, SCALE, SHIFT, 1, APRE);
      gemm(APRE, 160, W2b[li], ACT + c, CP, nullptr, rows, 16, 12);         // k3 -> +12 ch
      c += 12;
    }
    // transition: BN-ReLU-1x1 then avgpool(2)
    int Co = tCo[blk], Kp = a32(c);
    bnstats(ACT, CP, c, rows, F(tb[blk] + 1), F(tb[blk] + 0));
    k_bnrelu_bf16<<<ceilDiv((size_t)rows * Kp, 256), 256, 0, stream>>>(
        ACT, CP, c, Kp, rows, SCALE, SHIFT, APRE);
    gemm(APRE, Kp, Wtb[blk], TMP, Co, nullptr, rows, a16(Co), Co);
    int nextCP = (blk < 2) ? CPs[blk + 1] : 216;
    int rowsOut = rows / 2;
    k_avgpool2<<<ceilDiv((size_t)rowsOut * Co, 256), 256, 0, stream>>>(
        TMP, Co, Co, L, rowsOut, ACT, nextCP);
    rows = rowsOut; L >>= 1;
  }
  // ACT now holds z: rows = B*8 = 16384, ld 216 (VQ vectors along length dim)

  // ---- VQ + losses ----
  {
    int nvq = B * 216;
    int nb = ceilDiv((size_t)nvq, 256);
    k_vq<<<nb, 256, 0, stream>>>(ACT, 216, F(195), VQJ, LP);
    k_lossfin<<<1, 256, 0, stream>>>(LP, nb, 1.0f / (float)nvq, outF + 131072);
  }

  // ---- decoder ----
  k_decin<<<ceilDiv((size_t)32768 * 216, 256), 256, 0, stream>>>(F(195), VQJ, ACT, 216);
  // deconv3 (216->96, L=16), no BN before
  k_im2col3<<<ceilDiv((size_t)32768 * 672, 256), 256, 0, stream>>>(
      ACT, 216, 216, 16, 16, 672, 32768, SCALE, SHIFT, 0, APRE);
  gemm(APRE, 672, Wd[0], TMP, 96, F(190), 32768, 96, 96);
  bnstats(TMP, 96, 96, 32768, F(194), F(183));
  // upsample x2 + BN-ReLU + im2col, deconv2 (96->72, L=32)
  k_im2col3<<<ceilDiv((size_t)65536 * 288, 256), 256, 0, stream>>>(
      TMP, 96, 96, 32, 16, 288, 65536, SCALE, SHIFT, 1, APRE);
  gemm(APRE, 288, Wd[1], ACT, 72, F(188), 65536, 80, 72);
  bnstats(ACT, 72, 72, 65536, F(193), F(182));
  // upsample x2 + BN-ReLU + im2col, deconv1 (72->24, L=64)
  k_im2col3<<<ceilDiv((size_t)131072 * 224, 256), 256, 0, stream>>>(
      ACT, 72, 72, 64, 32, 224, 131072, SCALE, SHIFT, 1, APRE);
  gemm(APRE, 224, Wd[2], TMP, 24, F(186), 131072, 32, 24);
  bnstats(TMP, 24, 24, 131072, F(192), F(181));
  // BN-ReLU + im2col (no upsample), deconv0 (24->1, L=64) -> recon
  k_im2col3<<<ceilDiv((size_t)131072 * 96, 256), 256, 0, stream>>>(
      TMP, 24, 24, 64, 64, 96, 131072, SCALE, SHIFT, 1, APRE);
  gemm(APRE, 96, Wd[3], outF, 1, F(184), 131072, 16, 1);
}
